// QueryConditionedModel_35270271435179
// MI455X (gfx1250) — compile-verified, matrix operand order checked by reading, not run on hardware
//
#include <hip/hip_runtime.h>
#include <hip/hip_bf16.h>
#include <cstdint>
#include <cstddef>

// ---------------------------------------------------------------------------
// QueryConditionedModel for MI455X (gfx1250, wave32, WMMA).
// FF GEMMs: v_wmma_f32_16x16x32_bf16 with double-buffered
// global_load_async_to_lds_b128 staging of the B tile (ASYNCcnt path).
// Steady-state K-loop is branch-free (last step peeled); A fragment loads
// are hoisted ahead of the async wait + barrier to hide global latency.
// ---------------------------------------------------------------------------

typedef __attribute__((ext_vector_type(16))) __bf16 v16bf;
typedef __attribute__((ext_vector_type(8)))  __bf16 v8bf;
typedef __attribute__((ext_vector_type(8)))  float  v8f;

static constexpr int V_  = 32000;
static constexpr int H_  = 1024;
static constexpr int M_  = 64;
static constexpr int NT_ = 16;
static constexpr int B_  = 16;
static constexpr int T_  = 2048;
static constexpr int BT_ = B_ * T_;   // 32768
static constexpr int HP_ = H_ / 2;    // 512

static constexpr int LDS_PITCH = 80;  // 32 bf16 (64B) + 16B pad -> conflict-free ds_load_b128

// ------------------------- CDNA5 async helpers -----------------------------
__device__ __forceinline__ void async_b128(void* lds, const void* g) {
  unsigned l = (unsigned)(uintptr_t)lds;      // low 32 bits of flat LDS ptr = LDS offset
  asm volatile("global_load_async_to_lds_b128 %0, %1, off"
               :: "v"(l), "v"(g) : "memory");
}

template <int N>
__device__ __forceinline__ void wait_asynccnt() {
  asm volatile("s_wait_asynccnt %0" :: "i"(N) : "memory");
}

// ------------------------------ gather -------------------------------------
__global__ __launch_bounds__(256) void k_gather(const int* __restrict__ seq,
                                                const float* __restrict__ embed,
                                                float* __restrict__ h0) {
  int token = blockIdx.x;
  int row   = seq[token];
  const float4* src = (const float4*)(embed + (size_t)row * H_);
  float4*       dst = (float4*)(h0 + (size_t)token * H_);
  for (int i = threadIdx.x; i < H_ / 4; i += blockDim.x) dst[i] = src[i];
}

// -------------------- weight transpose + f32 -> bf16 -----------------------
// wt[n*K + k] = w[k*N + n]   (N-major so WMMA B-fragment loads are contiguous)
__global__ __launch_bounds__(256) void k_tconv(const float* __restrict__ w,
                                               __bf16* __restrict__ wt,
                                               int K, int N) {
  int idx = blockIdx.x * blockDim.x + threadIdx.x;
  if (idx < K * N) {
    int n = idx / K, k = idx % K;
    wt[idx] = (__bf16)w[(size_t)k * N + n];
  }
}

// ----------------------- WMMA fragment loaders -----------------------------
// 16-bit A 16x32 layout (ISA 7.12.2): lane<16 -> K base 0, lane>=16 -> K base 8.
// vector elems 0..7  <-> K = kk+kb+0..7 ; elems 8..15 <-> K = kk+16+kb+0..7
__device__ __forceinline__ v16bf frag_from_f32(const float* __restrict__ row,
                                               int kk, int kb) {
  const float* p0 = row + kk + kb;
  const float* p1 = row + kk + 16 + kb;
  v16bf r;
#pragma unroll
  for (int e = 0; e < 8; ++e) { r[e] = (__bf16)p0[e]; r[e + 8] = (__bf16)p1[e]; }
  return r;
}

__device__ __forceinline__ v16bf frag_from_bf16(const __bf16* __restrict__ row,
                                                int kk, int kb) {
  v8bf lo = *(const v8bf*)(row + kk + kb);
  v8bf hi = *(const v8bf*)(row + kk + 16 + kb);
  v16bf r;
#pragma unroll
  for (int e = 0; e < 8; ++e) { r[e] = lo[e]; r[e + 8] = hi[e]; }
  return r;
}

// B fragment from the LDS-staged tile: rowbase points at this n-row's 32 K
// values (64B + pad); kb2 = byte offset for this lane's K base.
__device__ __forceinline__ v16bf frag_from_lds(const char* rowbase, int kb2) {
  v8bf lo = *(const v8bf*)(rowbase + kb2);        // K = kk+kb .. +7
  v8bf hi = *(const v8bf*)(rowbase + 32 + kb2);   // K = kk+16+kb .. +7
  v16bf r;
#pragma unroll
  for (int e = 0; e < 8; ++e) { r[e] = lo[e]; r[e + 8] = hi[e]; }
  return r;
}

// ------------------------------ WMMA GEMM ----------------------------------
// C[m][n] = act( sum_k A[m][k] * Bt[n][k] + bias[n] )
// Block: 8 waves -> 64(M) x 128(N) tile; wave tile 16x64 (4 accumulators).
// B tile (128 x 32 bf16 = 8KB) double-buffered in LDS via async DMA.
template <bool A_BF16, bool RELU_BF16_OUT>
__global__ __launch_bounds__(256) void k_gemm(const void* __restrict__ Aptr,
                                              const __bf16* __restrict__ Bt,
                                              const float* __restrict__ bias,
                                              void* __restrict__ Cptr,
                                              int N, int K) {
  __shared__ __align__(16) char sB[2][128 * LDS_PITCH];   // 2 x 10KB

  const int tid  = threadIdx.x;
  const int lane = tid & 31;
  const int wave = tid >> 5;
  const int wm = wave & 3, wn = wave >> 2;
  const int m0   = blockIdx.y * 64 + wm * 16;
  const int nblk = blockIdx.x * 128;
  const int n0   = nblk + wn * 64;
  const int l15  = lane & 15;
  const int kb   = (lane & 16) ? 8 : 0;
  const int kb2  = kb * 2;

  // per-thread async-copy assignment: row = tid/2, 32B half = tid&1
  const int srow = tid >> 1, shalf = tid & 1;
  const __bf16* gsrc_row = Bt + (size_t)(nblk + srow) * K + shalf * 16;
  const int soff = srow * LDS_PITCH + shalf * 32;

  const float*  arow_f = A_BF16 ? nullptr : (const float*)Aptr + (size_t)(m0 + l15) * K;
  const __bf16* arow_b = A_BF16 ? (const __bf16*)Aptr + (size_t)(m0 + l15) * K : nullptr;

  const char* fbase0 = &sB[0][0] + (size_t)(wn * 64 + l15) * LDS_PITCH;
  const char* fbase1 = &sB[1][0] + (size_t)(wn * 64 + l15) * LDS_PITCH;

  // prologue: stage K-step 0 into buffer 0
  async_b128(&sB[0][0] + soff,      gsrc_row);
  async_b128(&sB[0][0] + soff + 16, (const char*)gsrc_row + 16);

  v8f c0 = {}, c1 = {}, c2 = {}, c3 = {};
  int cur = 0;
  int kk = 0;
  // steady state: branch-free; stage kk+32 while computing kk
  for (; kk < K - 32; kk += 32) {
    // hoist A fragment load (global + cvt) ahead of async wait + barrier
    v16bf a = A_BF16 ? frag_from_bf16(arow_b, kk, kb)
                     : frag_from_f32(arow_f, kk, kb);

    char* l = (cur ? &sB[0][0] : &sB[1][0]) + soff;   // next buffer
    const __bf16* g = gsrc_row + (kk + 32);
    async_b128(l,      g);
    async_b128(l + 16, (const char*)g + 16);
    wait_asynccnt<2>();                               // step-kk copies retired
    __syncthreads();                                  // all waves' copies visible

    const char* base = cur ? fbase1 : fbase0;
    v16bf b0 = frag_from_lds(base,                  kb2);
    v16bf b1 = frag_from_lds(base + 16 * LDS_PITCH, kb2);
    v16bf b2 = frag_from_lds(base + 32 * LDS_PITCH, kb2);
    v16bf b3 = frag_from_lds(base + 48 * LDS_PITCH, kb2);
    c0 = __builtin_amdgcn_wmma_f32_16x16x32_bf16(false, a, false, b0, (short)0, c0, false, false);
    c1 = __builtin_amdgcn_wmma_f32_16x16x32_bf16(false, a, false, b1, (short)0, c1, false, false);
    c2 = __builtin_amdgcn_wmma_f32_16x16x32_bf16(false, a, false, b2, (short)0, c2, false, false);
    c3 = __builtin_amdgcn_wmma_f32_16x16x32_bf16(false, a, false, b3, (short)0, c3, false, false);

    __syncthreads();                                  // done reading sB[cur]
    cur ^= 1;
  }
  // epilogue: last K-step (no further staging)
  {
    v16bf a = A_BF16 ? frag_from_bf16(arow_b, kk, kb)
                     : frag_from_f32(arow_f, kk, kb);
    wait_asynccnt<0>();
    __syncthreads();
    const char* base = cur ? fbase1 : fbase0;
    v16bf b0 = frag_from_lds(base,                  kb2);
    v16bf b1 = frag_from_lds(base + 16 * LDS_PITCH, kb2);
    v16bf b2 = frag_from_lds(base + 32 * LDS_PITCH, kb2);
    v16bf b3 = frag_from_lds(base + 48 * LDS_PITCH, kb2);
    c0 = __builtin_amdgcn_wmma_f32_16x16x32_bf16(false, a, false, b0, (short)0, c0, false, false);
    c1 = __builtin_amdgcn_wmma_f32_16x16x32_bf16(false, a, false, b1, (short)0, c1, false, false);
    c2 = __builtin_amdgcn_wmma_f32_16x16x32_bf16(false, a, false, b2, (short)0, c2, false, false);
    c3 = __builtin_amdgcn_wmma_f32_16x16x32_bf16(false, a, false, b3, (short)0, c3, false, false);
  }

  const int mbase = m0 + ((lane & 16) ? 8 : 0);
  const int n = n0 + l15;
#pragma unroll
  for (int r = 0; r < 8; ++r) {
    size_t ro = (size_t)(mbase + r) * N;
    if constexpr (RELU_BF16_OUT) {
      __bf16* C = (__bf16*)Cptr;
      C[ro + n]      = (__bf16)fmaxf(c0[r] + bias[n],      0.f);
      C[ro + n + 16] = (__bf16)fmaxf(c1[r] + bias[n + 16], 0.f);
      C[ro + n + 32] = (__bf16)fmaxf(c2[r] + bias[n + 32], 0.f);
      C[ro + n + 48] = (__bf16)fmaxf(c3[r] + bias[n + 48], 0.f);
    } else {
      float* C = (float*)Cptr;
      C[ro + n]      = c0[r] + bias[n];
      C[ro + n + 16] = c1[r] + bias[n + 16];
      C[ro + n + 32] = c2[r] + bias[n + 32];
      C[ro + n + 48] = c3[r] + bias[n + 48];
    }
  }
}

// --------------------- residual + LayerNorm (in place) ---------------------
__global__ __launch_bounds__(256) void k_ln(float* __restrict__ hbuf,
                                            const float* __restrict__ ff,
                                            const float* __restrict__ g,
                                            const float* __restrict__ bta) {
  int row = blockIdx.x, tid = threadIdx.x;
  float* hr       = hbuf + (size_t)row * H_;
  const float* fr = ff   + (size_t)row * H_;
  __shared__ float red[256];
  float x[4]; float s = 0.f;
#pragma unroll
  for (int i = 0; i < 4; ++i) { int idx = tid + i * 256; x[i] = hr[idx] + fr[idx]; s += x[i]; }
  red[tid] = s; __syncthreads();
  for (int st = 128; st > 0; st >>= 1) { if (tid < st) red[tid] += red[tid + st]; __syncthreads(); }
  float mu = red[0] / H_;
  __syncthreads();
  float v = 0.f;
#pragma unroll
  for (int i = 0; i < 4; ++i) { float d = x[i] - mu; v += d * d; }
  red[tid] = v; __syncthreads();
  for (int st = 128; st > 0; st >>= 1) { if (tid < st) red[tid] += red[tid + st]; __syncthreads(); }
  float rs = rsqrtf(red[0] / H_ + 1e-5f);
#pragma unroll
  for (int i = 0; i < 4; ++i) { int idx = tid + i * 256; hr[idx] = (x[i] - mu) * rs * g[idx] + bta[idx]; }
}

// ----------------------------- mean over T ---------------------------------
__global__ __launch_bounds__(256) void k_meanh(const float* __restrict__ h,
                                               float* __restrict__ mean_h) {
  int j = blockIdx.x * 256 + threadIdx.x;
  int b = blockIdx.y;
  float s = 0.f;
  for (int t = 0; t < T_; ++t) s += h[((size_t)b * T_ + t) * H_ + j];
  mean_h[b * H_ + j] = s / (float)T_;
}

// -------- type head: softmax(mean_h@type_w+b), tproj, gate constant --------
__global__ __launch_bounds__(256) void k_typehead(const float* __restrict__ mean_h,
                                                  const float* __restrict__ type_w,
                                                  const float* __restrict__ type_b,
                                                  const float* __restrict__ tproj_w,
                                                  const float* __restrict__ tproj_b,
                                                  const float* __restrict__ gate_w,
                                                  const float* __restrict__ gate_b,
                                                  float* __restrict__ type_dist_out,
                                                  float* __restrict__ gcon) {
  int b = blockIdx.x, tid = threadIdx.x;
  __shared__ float part[256];
  __shared__ float dist[NT_];
  __shared__ float te[HP_];

  int c = tid & 15, seg = tid >> 4;      // 16 logits x 16 k-segments of 64
  float p = 0.f;
  for (int k = seg * 64; k < seg * 64 + 64; ++k)
    p += mean_h[b * H_ + k] * type_w[k * NT_ + c];
  part[tid] = p; __syncthreads();
  if (tid < NT_) {
    float s = type_b[tid];
    for (int g = 0; g < 16; ++g) s += part[g * 16 + tid];
    dist[tid] = s;
  }
  __syncthreads();
  if (tid == 0) {
    float mx = dist[0];
    for (int i = 1; i < NT_; ++i) mx = fmaxf(mx, dist[i]);
    float ss = 0.f;
    for (int i = 0; i < NT_; ++i) { dist[i] = expf(dist[i] - mx); ss += dist[i]; }
    for (int i = 0; i < NT_; ++i) dist[i] /= ss;
  }
  __syncthreads();
  if (tid < NT_) type_dist_out[b * NT_ + tid] = dist[tid];

  for (int j = tid; j < HP_; j += 256) {
    float s = tproj_b[j];
    for (int cc = 0; cc < NT_; ++cc) s += dist[cc] * tproj_w[cc * HP_ + j];
    te[j] = s;
  }
  __syncthreads();
  float gp = 0.f;
  for (int j = tid; j < HP_; j += 256) gp += te[j] * gate_w[H_ + j];
  part[tid] = gp; __syncthreads();
  for (int st = 128; st > 0; st >>= 1) { if (tid < st) part[tid] += part[tid + st]; __syncthreads(); }
  if (tid == 0) gcon[b] = part[0] + gate_b[0];
}

// ------------------------------- gate --------------------------------------
__global__ __launch_bounds__(256) void k_gate(const float* __restrict__ h,
                                              const float* __restrict__ gate_w,
                                              const float* __restrict__ gcon,
                                              float* __restrict__ gate) {
  int token = blockIdx.x, tid = threadIdx.x;
  int b = token / T_;
  const float* hr = h + (size_t)token * H_;
  float p = 0.f;
  for (int k = tid; k < H_; k += 256) p += hr[k] * gate_w[k];
  __shared__ float part[256];
  part[tid] = p; __syncthreads();
  for (int st = 128; st > 0; st >>= 1) { if (tid < st) part[tid] += part[tid + st]; __syncthreads(); }
  if (tid == 0) {
    float x = part[0] + gcon[b];
    gate[token] = 1.f / (1.f + expf(-x));
  }
}

// --------------------- deterministic iterative top-64 ----------------------
__global__ __launch_bounds__(256) void k_topk(const float* __restrict__ gate,
                                              int* __restrict__ topidx) {
  int b = blockIdx.x, tid = threadIdx.x;
  __shared__ float sg[T_];
  __shared__ float rv[256];
  __shared__ int   ri[256];
  for (int t = tid; t < T_; t += 256) sg[t] = gate[b * T_ + t];
  __syncthreads();
  for (int m = 0; m < M_; ++m) {
    float bv = -3.4e38f; int bi = T_;
    for (int t = tid; t < T_; t += 256) { float v = sg[t]; if (v > bv) { bv = v; bi = t; } }
    rv[tid] = bv; ri[tid] = bi; __syncthreads();
    for (int st = 128; st > 0; st >>= 1) {
      if (tid < st) {
        if (rv[tid + st] > rv[tid] || (rv[tid + st] == rv[tid] && ri[tid + st] < ri[tid])) {
          rv[tid] = rv[tid + st]; ri[tid] = ri[tid + st];
        }
      }
      __syncthreads();
    }
    if (tid == 0) { topidx[b * M_ + m] = ri[0]; sg[ri[0]] = -3.4e38f; }
    __syncthreads();
  }
}

// ---------------------------- q projection ---------------------------------
__global__ __launch_bounds__(256) void k_qproj(const float* __restrict__ h,
                                               const float* __restrict__ qw,
                                               const float* __restrict__ qb,
                                               float* __restrict__ q) {
  int j = blockIdx.x * 256 + threadIdx.x;
  int b = blockIdx.y;
  const float* hr = h + ((size_t)b * T_ + (T_ - 1)) * H_;
  float acc = qb[j];
  for (int k = 0; k < H_; ++k) acc += hr[k] * qw[(size_t)k * H_ + j];
  q[b * H_ + j] = acc;
}

// ------------------- attention over the 64 selected rows -------------------
__global__ __launch_bounds__(256) void k_attn(const float* __restrict__ h,
                                              const int* __restrict__ topidx,
                                              const float* __restrict__ q,
                                              float* __restrict__ ctx) {
  int b = blockIdx.x, tid = threadIdx.x;
  __shared__ float sp[256];
  __shared__ float attn[M_];
  int m = tid >> 2, g = tid & 3;              // 64 rows x 4 partial groups
  int row = topidx[b * M_ + m];
  const float* hr = h + ((size_t)b * T_ + row) * H_;
  const float* qb = q + b * H_;
  float p = 0.f;
  for (int k = g * (H_ / 4); k < (g + 1) * (H_ / 4); ++k) p += hr[k] * qb[k];
  sp[tid] = p; __syncthreads();
  if (g == 0) attn[m] = sp[tid] + sp[tid + 1] + sp[tid + 2] + sp[tid + 3];
  __syncthreads();
  if (tid == 0) {
    float mx = attn[0];
    for (int i = 1; i < M_; ++i) mx = fmaxf(mx, attn[i]);
    float s = 0.f;
    for (int i = 0; i < M_; ++i) { attn[i] = expf(attn[i] - mx); s += attn[i]; }
    for (int i = 0; i < M_; ++i) attn[i] /= s;
  }
  __syncthreads();
  for (int j = tid; j < H_; j += 256) {
    float s = 0.f;
    for (int mm = 0; mm < M_; ++mm) {
      int r2 = topidx[b * M_ + mm];
      s += attn[mm] * h[((size_t)b * T_ + r2) * H_ + j];
    }
    ctx[b * H_ + j] = s;
  }
}

// ------------------------------- logits ------------------------------------
__global__ __launch_bounds__(256) void k_logits(const float* __restrict__ ctx,
                                                const float* __restrict__ out_w,
                                                const float* __restrict__ out_b,
                                                float* __restrict__ out) {
  int v = blockIdx.x * 256 + threadIdx.x;
  __shared__ float sctx[B_ * H_];             // 64 KB LDS
  for (int i = threadIdx.x; i < B_ * H_; i += 256) sctx[i] = ctx[i];
  __syncthreads();
  float acc[B_];
#pragma unroll
  for (int b = 0; b < B_; ++b) acc[b] = out_b[v];
  for (int k = 0; k < H_; ++k) {
    float w = out_w[(size_t)k * V_ + v];
#pragma unroll
    for (int b = 0; b < B_; ++b) acc[b] += sctx[b * H_ + k] * w;
  }
#pragma unroll
  for (int b = 0; b < B_; ++b) out[(size_t)b * V_ + v] = acc[b];
}

// ---------------------------------------------------------------------------
extern "C" void kernel_launch(void* const* d_in, const int* in_sizes, int n_in,
                              void* d_out, int out_size, void* d_ws, size_t ws_size,
                              hipStream_t stream) {
  const int*   seq     = (const int*)  d_in[0];
  const float* embed   = (const float*)d_in[1];
  const float* ff1_w   = (const float*)d_in[2];
  const float* ff1_b   = (const float*)d_in[3];
  const float* ff2_w   = (const float*)d_in[4];
  const float* ff2_b   = (const float*)d_in[5];
  const float* ln_g    = (const float*)d_in[6];
  const float* ln_b    = (const float*)d_in[7];
  const float* type_w  = (const float*)d_in[8];
  const float* type_b  = (const float*)d_in[9];
  const float* tproj_w = (const float*)d_in[10];
  const float* tproj_b = (const float*)d_in[11];
  const float* gate_w  = (const float*)d_in[12];
  const float* gate_b  = (const float*)d_in[13];
  const float* q_w     = (const float*)d_in[14];
  const float* q_b     = (const float*)d_in[15];
  const float* out_w   = (const float*)d_in[16];
  const float* out_b   = (const float*)d_in[17];

  float* out = (float*)d_out;                 // logits [B,V] then type_dist [B,NT]
  float* type_dist_out = out + (size_t)B_ * V_;

  // ---- carve workspace ----
  char* ws = (char*)d_ws;
  auto carve = [&](size_t bytes) -> void* {
    void* p = (void*)ws;
    ws += (bytes + 255) & ~(size_t)255;
    return p;
  };
  float*  hbuf  = (float*) carve((size_t)BT_ * H_ * 4);       // h0, then h (in place)
  float*  ffb   = (float*) carve((size_t)BT_ * H_ * 4);       // ff output
  __bf16* mid   = (__bf16*)carve((size_t)BT_ * 2 * H_ * 2);   // relu(ff1) bf16
  __bf16* ff1t  = (__bf16*)carve((size_t)2 * H_ * H_ * 2);    // [2H][H]
  __bf16* ff2t  = (__bf16*)carve((size_t)2 * H_ * H_ * 2);    // [H][2H]
  float*  meanh = (float*) carve((size_t)B_ * H_ * 4);
  float*  gcon  = (float*) carve((size_t)B_ * 4);
  float*  gate  = (float*) carve((size_t)BT_ * 4);
  int*    tidx  = (int*)   carve((size_t)B_ * M_ * 4);
  float*  qv    = (float*) carve((size_t)B_ * H_ * 4);
  float*  ctx   = (float*) carve((size_t)B_ * H_ * 4);

  // 1. embedding gather
  k_gather<<<BT_, 256, 0, stream>>>(seq, embed, hbuf);

  // 2. weight transpose/convert to bf16 (N-major for WMMA B fragments)
  k_tconv<<<(2 * H_ * H_) / 256, 256, 0, stream>>>(ff1_w, ff1t, H_, 2 * H_);
  k_tconv<<<(2 * H_ * H_) / 256, 256, 0, stream>>>(ff2_w, ff2t, 2 * H_, H_);

  // 3. GEMM1: mid = relu(h0 @ ff1_w + b1)   [32768 x 2048], K=1024
  k_gemm<false, true><<<dim3((2 * H_) / 128, BT_ / 64), 256, 0, stream>>>(
      hbuf, ff1t, ff1_b, mid, 2 * H_, H_);

  // 4. GEMM2: ff = mid @ ff2_w + b2         [32768 x 1024], K=2048
  k_gemm<true, false><<<dim3(H_ / 128, BT_ / 64), 256, 0, stream>>>(
      mid, ff2t, ff2_b, ffb, H_, 2 * H_);

  // 5. h = LN(h0 + ff) (in place in hbuf)
  k_ln<<<BT_, 256, 0, stream>>>(hbuf, ffb, ln_g, ln_b);

  // 6. mean over T
  k_meanh<<<dim3(H_ / 256, B_), 256, 0, stream>>>(hbuf, meanh);

  // 7. type head (writes type_dist to output tail, gate constant to ws)
  k_typehead<<<B_, 256, 0, stream>>>(meanh, type_w, type_b, tproj_w, tproj_b,
                                     gate_w, gate_b, type_dist_out, gcon);

  // 8. gate sigmoid per token
  k_gate<<<BT_, 256, 0, stream>>>(hbuf, gate_w, gcon, gate);

  // 9. top-64 per batch (deterministic, smallest-index tie-break)
  k_topk<<<B_, 256, 0, stream>>>(gate, tidx);

  // 10. q projection (last token)
  k_qproj<<<dim3(H_ / 256, B_), 256, 0, stream>>>(hbuf, q_w, q_b, qv);

  // 11. attention over selected memory rows
  k_attn<<<B_, 256, 0, stream>>>(hbuf, tidx, qv, ctx);

  // 12. output logits
  k_logits<<<V_ / 256, 256, 0, stream>>>(ctx, out_w, out_b, out);

  (void)in_sizes; (void)n_in; (void)out_size; (void)ws_size;
}
